// DeformConv2d_60335700574627
// MI455X (gfx1250) — matrix operand order repaired and checked by reading, not compiled, not run on hardware
//
#include <hip/hip_runtime.h>
#include <stdint.h>

// ---------------------------------------------------------------------------
// B=8, C=64, H=W=128, OUTC=64, KS=3, N=9.
// Both convs are implicit GEMMs with K = n*64 + c  (n = kh*3+kw):
//   offsets : [18(pad 32) x 576] x [576 x 131072]   (integer taps)
//   main    : [64        x 576] x [576 x 131072]   (bilinear taps)
// x is converted ONCE to a zero-PADDED NHWC bf16 tensor xp[8][130][130][64]
// (17.3MB, L2-resident). All taps/corners live in [0,129] (exactly the
// reference's clipped padded coordinates), so every gather is an
// unconditional b128 load: no predication, no flat path, no cndmask.
// ---------------------------------------------------------------------------
#define B_    8
#define C_    64
#define H_    128
#define W_    128
#define HP_   130
#define WP_   130
#define OUTC_ 64
#define N_    9
#define KTOT  (C_ * N_)     // 576
#define NKC   (KTOT / 32)   // 18
#define PIX   (B_ * H_ * W_)
#define PIXP  (B_ * HP_ * WP_)   // 135200 padded pixels

#define WFRAG_DWORDS  (NKC * 4 * 32 * 8)   // main-conv A frags: 18432 dw = 72KB
#define WPFRAG_DWORDS (NKC * 2 * 32 * 8)   // offset-conv A frags: 9216 dw = 36KB

typedef __attribute__((ext_vector_type(16))) __bf16 v16bf;
typedef __attribute__((ext_vector_type(8)))  float  v8f;
typedef unsigned int u32x4_t __attribute__((ext_vector_type(4)));
typedef int          i32x8_t __attribute__((ext_vector_type(8)));
typedef int          i32x4_t __attribute__((ext_vector_type(4)));

__device__ __forceinline__ uint16_t f2bf(float f) {
    uint32_t u = __float_as_uint(f);
    uint32_t r = u + 0x7FFFu + ((u >> 16) & 1u);
    return (uint16_t)(r >> 16);
}
__device__ __forceinline__ uint32_t bfdup(float f) {  // bf16 value in both halves
    uint32_t h = f2bf(f);
    return h | (h << 16);
}
// packed bf16 math (VOP3P): blend two channels per instruction
__device__ __forceinline__ uint32_t pk_mul_bf16(uint32_t a, uint32_t b) {
    uint32_t d;
    asm volatile("v_pk_mul_bf16 %0, %1, %2" : "=v"(d) : "v"(a), "v"(b));
    return d;
}
__device__ __forceinline__ uint32_t pk_fma_bf16(uint32_t a, uint32_t b, uint32_t c) {
    uint32_t d;
    asm volatile("v_pk_fma_bf16 %0, %1, %2, %3" : "=v"(d) : "v"(a), "v"(b), "v"(c));
    return d;
}

// TDM: DMA `ndwords` (<=65535) contiguous dwords Global -> LDS offset 0.
__device__ __forceinline__ void tdm_load_lds(const void* gptr, uint32_t nd) {
    uint64_t ga = (uint64_t)(uintptr_t)gptr;
    u32x4_t g0;
    g0[0] = 1u;                                           // count=1
    g0[1] = 0u;                                           // lds_addr = dynamic base
    g0[2] = (uint32_t)ga;
    g0[3] = (uint32_t)((ga >> 32) & 0x1FFFFFFu) | (2u << 30); // type=2
    i32x8_t g1;
    g1[0] = 0x00020000;                       // mask=0, data_size=4B
    g1[1] = (int)((nd & 0xFFFFu) << 16);      // tensor_dim0 lo16
    g1[2] = (int)((nd >> 16) | (1u << 16));   // tensor_dim0 hi16 | tensor_dim1=1
    g1[3] = (int)((nd & 0xFFFFu) << 16);      // tile_dim0
    g1[4] = 1;                                // tile_dim1=1
    g1[5] = (int)nd;                          // tensor_dim0_stride
    g1[6] = 0; g1[7] = 0;
    i32x4_t gz = {0, 0, 0, 0};
#if defined(__clang_major__) && (__clang_major__ >= 23)
    i32x8_t gz8 = {0, 0, 0, 0, 0, 0, 0, 0};
    __builtin_amdgcn_tensor_load_to_lds(g0, g1, gz, gz, gz8, 0);
#else
    __builtin_amdgcn_tensor_load_to_lds(g0, g1, gz, gz, 0);
#endif
    __builtin_amdgcn_s_wait_tensorcnt(0);
}

union AFrag { uint4 q[2]; v16bf v; };
union BFrag { uint4 q[2]; uint32_t u[8]; v16bf v; };

// ---------------------------------------------------------------------------
// Kernel: NCHW f32 -> zero-padded NHWC bf16 (packed pairs).
// One thread per PADDED pixel; border threads store zeros.
// ---------------------------------------------------------------------------
__global__ void pad_nhwc_bf16(const float* __restrict__ x, uint32_t* __restrict__ xp) {
    int p = blockIdx.x * 256 + threadIdx.x;
    if (p >= PIXP) return;
    int wc = p % WP_;
    int t  = p / WP_;
    int hr = t % HP_;
    int b  = t / HP_;
    uint4* dst = (uint4*)(xp + (size_t)p * 32);
    if (hr == 0 || hr == HP_ - 1 || wc == 0 || wc == WP_ - 1) {
        uint4 z = {0, 0, 0, 0};
        #pragma unroll
        for (int q = 0; q < 8; ++q) dst[q] = z;
        return;
    }
    const float* xb = x + (size_t)b * C_ * H_ * W_ + (hr - 1) * W_ + (wc - 1);
    uint32_t v[32];
    #pragma unroll
    for (int c2 = 0; c2 < 32; ++c2) {
        float lo = xb[(size_t)(2 * c2) * (H_ * W_)];
        float hi = xb[(size_t)(2 * c2 + 1) * (H_ * W_)];
        v[c2] = (uint32_t)f2bf(lo) | ((uint32_t)f2bf(hi) << 16);
    }
    #pragma unroll
    for (int q = 0; q < 8; ++q) dst[q] = ((const uint4*)v)[q];
}

// ---------------------------------------------------------------------------
// A-fragment packers: wave32 16x32 bf16 A, contiguous k order used here (B matches):
//   lanes 0-15 (M=lane):   V0..V7 -> K=0..15
//   lanes 16-31 (M=l-16):  V0..V7 -> K=16..31
// K = kc*32 + klocal; n = kc>>1; c = (kc&1)*32 + klocal.
// ---------------------------------------------------------------------------
__device__ __forceinline__ int a_kbase(int lane, int vp) {
    if (lane < 16) return 2 * vp;
    return 16 + 2 * vp;
}

__global__ void prep_w(const float* __restrict__ w_conv, uint32_t* __restrict__ wfrag) {
    int idx = blockIdx.x * 256 + threadIdx.x;
    if (idx >= WFRAG_DWORDS) return;
    int vp = idx & 7, lane = (idx >> 3) & 31, tile = (idx >> 8) & 3, kc = idx >> 10;
    int oc = tile * 16 + (lane & 15);
    int n  = kc >> 1;
    int c0 = ((kc & 1) << 5) + a_kbase(lane, vp);
    uint16_t lo = f2bf(w_conv[oc * KTOT + c0 * N_ + n]);
    uint16_t hi = f2bf(w_conv[oc * KTOT + (c0 + 1) * N_ + n]);
    wfrag[idx] = (uint32_t)lo | ((uint32_t)hi << 16);
}

__global__ void prep_wp(const float* __restrict__ w_pconv, uint32_t* __restrict__ wpfrag) {
    int idx = blockIdx.x * 256 + threadIdx.x;
    if (idx >= WPFRAG_DWORDS) return;
    int vp = idx & 7, lane = (idx >> 3) & 31, tile = (idx >> 8) & 1, kc = idx >> 9;
    int oc = tile * 16 + (lane & 15);
    uint32_t packed = 0;
    if (oc < 18) {   // pad 18 -> 32 rows with zeros
        int n  = kc >> 1;
        int c0 = ((kc & 1) << 5) + a_kbase(lane, vp);
        uint16_t lo = f2bf(w_pconv[oc * KTOT + c0 * N_ + n]);
        uint16_t hi = f2bf(w_pconv[oc * KTOT + (c0 + 1) * N_ + n]);
        packed = (uint32_t)lo | ((uint32_t)hi << 16);
    }
    wpfrag[idx] = packed;
}

// ---------------------------------------------------------------------------
// Offset conv as WMMA GEMM: [32(18) oc x 576] x [576 x 16 px] per wave.
// Taps at padded coords (hr+kh, wc+kw) in [0,129] -> unconditional b128 loads.
// ---------------------------------------------------------------------------
__global__ void offset_gemm(const uint32_t* __restrict__ xp,
                            const uint32_t* __restrict__ wpfrag,
                            const float* __restrict__ b_pconv,
                            float* __restrict__ offs) {
    extern __shared__ uint32_t ldsA[];
    const int tid = threadIdx.x, lane = tid & 31, wave = tid >> 5;
    if (wave == 0) tdm_load_lds(wpfrag, WPFRAG_DWORDS);
    __syncthreads();

    const int px = lane & 15, khalf = lane >> 4;
    int pix = (blockIdx.x * 8 + wave) * 16 + px;
    int wc = pix & (W_ - 1), hr = (pix >> 7) & (H_ - 1), b = pix >> 14;
    const uint4* xp4 = (const uint4*)xp;   // 8 x uint4 per padded pixel

    v8f acc[2];
    #pragma unroll
    for (int t = 0; t < 2; ++t) acc[t] = (v8f){0,0,0,0,0,0,0,0};

    #pragma unroll
    for (int n = 0; n < N_; ++n) {
        int tap = ((b * HP_ + hr + n / 3) * WP_ + wc + n % 3) * 8;  // uint4 index
        #pragma unroll
        for (int half = 0; half < 2; ++half) {
            const int kc = 2 * n + half;
            const int q  = half * 4 + khalf * 2;
            BFrag bu;
            bu.q[0] = xp4[tap + q];
            bu.q[1] = xp4[tap + q + 1];
            const uint4* ap = (const uint4*)ldsA + ((size_t)(kc * 2) * 32 + lane) * 2;
            AFrag a0, a1;
            a0.q[0] = ap[0];  a0.q[1] = ap[1];
            a1.q[0] = ap[64]; a1.q[1] = ap[65];
            acc[0] = __builtin_amdgcn_wmma_f32_16x16x32_bf16(
                false, a0.v, false, bu.v, (short)0, acc[0], false, false);
            acc[1] = __builtin_amdgcn_wmma_f32_16x16x32_bf16(
                false, a1.v, false, bu.v, (short)0, acc[1], false, false);
        }
    }
    // keep oc < 18, add bias; offsets layout [b][18][h][w]
    #pragma unroll
    for (int t = 0; t < 2; ++t)
        #pragma unroll
        for (int r = 0; r < 8; ++r) {
            int oc = t * 16 + khalf * 8 + r;
            if (oc < 18)
                offs[((b * 18 + oc) * H_ + hr) * W_ + wc] = acc[t][r] + b_pconv[oc];
        }
}

// ---------------------------------------------------------------------------
// Main deformable GEMM: per wave 16 px x 64 oc (4 tiles). Per sampling point:
// corners are clipped padded coords in [0,129] -> 4 x 2 unconditional b128
// gathers, packed-bf16 bilinear blend, 4 back-to-back WMMAs per K-chunk.
// ---------------------------------------------------------------------------
__global__ void deform_gemm(const uint32_t* __restrict__ xp,
                            const float* __restrict__ offs,
                            const uint32_t* __restrict__ wfrag,
                            float* __restrict__ out) {
    extern __shared__ uint32_t ldsA[];
    const int tid = threadIdx.x, lane = tid & 31, wave = tid >> 5;
    if (wave == 0) tdm_load_lds(wfrag, WFRAG_DWORDS);
    __syncthreads();

    const int px = lane & 15, khalf = lane >> 4;
    int pix = (blockIdx.x * 8 + wave) * 16 + px;
    int wc = pix & (W_ - 1), hr = (pix >> 7) & (H_ - 1), b = pix >> 14;
    const uint4* xp4 = (const uint4*)xp;

    v8f acc[4];
    #pragma unroll
    for (int t = 0; t < 4; ++t) acc[t] = (v8f){0,0,0,0,0,0,0,0};

    #pragma unroll
    for (int n = 0; n < N_; ++n) {
        // ---- sampling coords (reference math; padded coords in [0,129]) ----
        float offx = offs[((b * 18 + n) * H_ + hr) * W_ + wc];
        float offy = offs[((b * 18 + 9 + n) * H_ + hr) * W_ + wc];
        float pxf = (float)(hr + 1) + (float)(n / 3 - 1) + offx;
        float pyf = (float)(wc + 1) + (float)(n % 3 - 1) + offy;
        float fx = floorf(pxf), fy = floorf(pyf);
        float qltx = fminf(fmaxf(fx,       0.f), 129.f);
        float qlty = fminf(fmaxf(fy,       0.f), 129.f);
        float qrbx = fminf(fmaxf(fx + 1.f, 0.f), 129.f);
        float qrby = fminf(fmaxf(fy + 1.f, 0.f), 129.f);
        float pxc  = fminf(fmaxf(pxf,      0.f), 129.f);
        float pyc  = fminf(fmaxf(pyf,      0.f), 129.f);
        uint32_t glt = bfdup((1.f + (qltx - pxc)) * (1.f + (qlty - pyc)));
        uint32_t grb = bfdup((1.f - (qrbx - pxc)) * (1.f - (qrby - pyc)));
        uint32_t glb = bfdup((1.f + (qltx - pxc)) * (1.f - (qrby - pyc)));
        uint32_t grt = bfdup((1.f - (qrbx - pxc)) * (1.f + (qlty - pyc)));

        // corner pixel indices in the PADDED tensor (always in-bounds)
        int ix_l = (int)qltx, iy_l = (int)qlty;
        int ix_r = (int)qrbx, iy_r = (int)qrby;
        int rb_l = (b * HP_ + ix_l) * WP_;
        int rb_r = (b * HP_ + ix_r) * WP_;
        int t_lt = (rb_l + iy_l) * 8;
        int t_rb = (rb_r + iy_r) * 8;
        int t_lb = (rb_l + iy_r) * 8;
        int t_rt = (rb_r + iy_l) * 8;

        #pragma unroll
        for (int half = 0; half < 2; ++half) {
            const int kc = 2 * n + half;
            const int q  = half * 4 + khalf * 2;     // uint4 index of 16-ch slice
            union { uint4 q[2]; uint32_t u[8]; } lt, rb, lb, rt;
            lt.q[0] = xp4[t_lt + q];  lt.q[1] = xp4[t_lt + q + 1];
            rb.q[0] = xp4[t_rb + q];  rb.q[1] = xp4[t_rb + q + 1];
            lb.q[0] = xp4[t_lb + q];  lb.q[1] = xp4[t_lb + q + 1];
            rt.q[0] = xp4[t_rt + q];  rt.q[1] = xp4[t_rt + q + 1];

            // packed-bf16 bilinear blend: 2 channels / instruction
            BFrag bu;
            #pragma unroll
            for (int j = 0; j < 8; ++j) {
                uint32_t v = pk_mul_bf16(glt, lt.u[j]);
                v = pk_fma_bf16(grb, rb.u[j], v);
                v = pk_fma_bf16(glb, lb.u[j], v);
                v = pk_fma_bf16(grt, rt.u[j], v);
                bu.u[j] = v;
            }

            // load all 4 A fragments (b128), then 4 back-to-back WMMAs
            const uint4* ap = (const uint4*)ldsA + ((size_t)(kc * 4) * 32 + lane) * 2;
            AFrag a[4];
            #pragma unroll
            for (int t = 0; t < 4; ++t) {
                a[t].q[0] = ap[t * 64];
                a[t].q[1] = ap[t * 64 + 1];
            }
            #pragma unroll
            for (int t = 0; t < 4; ++t)
                acc[t] = __builtin_amdgcn_wmma_f32_16x16x32_bf16(
                    false, a[t].v, false, bu.v, (short)0, acc[t], false, false);
        }
    }

    // D: lane column = px, rows = khalf*8 + r; out is NCHW f32
    #pragma unroll
    for (int t = 0; t < 4; ++t)
        #pragma unroll
        for (int r = 0; r < 8; ++r) {
            int oc = t * 16 + khalf * 8 + r;
            out[(((size_t)b * OUTC_ + oc) * H_ + hr) * W_ + wc] = acc[t][r];
        }
}

// ---------------------------------------------------------------------------
extern "C" void kernel_launch(void* const* d_in, const int* in_sizes, int n_in,
                              void* d_out, int out_size, void* d_ws, size_t ws_size,
                              hipStream_t stream) {
    const float* x       = (const float*)d_in[0];
    const float* w_pconv = (const float*)d_in[1];
    const float* b_pconv = (const float*)d_in[2];
    const float* w_conv  = (const float*)d_in[3];
    float* out = (float*)d_out;

    // workspace layout:
    //   [0, 17.3MB)  xp   : zero-padded NHWC bf16 (PIXP * 32 dwords)
    //   then         offs : f32 [B][18][H][W] (9.4MB)
    //   then         wfrag (72KB), wpfrag (36KB)
    char* ws = (char*)d_ws;
    uint32_t* xpb    = (uint32_t*)ws;
    float*    offs   = (float*)(ws + (size_t)PIXP * 32 * 4);
    uint32_t* wfrag  = (uint32_t*)(ws + (size_t)PIXP * 32 * 4 + (size_t)B_ * 18 * H_ * W_ * 4);
    uint32_t* wpfrag = wfrag + WFRAG_DWORDS;

    pad_nhwc_bf16<<<(PIXP + 255) / 256, 256, 0, stream>>>(x, xpb);
    prep_w <<<(WFRAG_DWORDS  + 255) / 256, 256, 0, stream>>>(w_conv, wfrag);
    prep_wp<<<(WPFRAG_DWORDS + 255) / 256, 256, 0, stream>>>(w_pconv, wpfrag);
    offset_gemm<<<1024, 256, WPFRAG_DWORDS * sizeof(uint32_t), stream>>>(xpb, wpfrag, b_pconv, offs);
    deform_gemm<<<1024, 256, WFRAG_DWORDS  * sizeof(uint32_t), stream>>>(xpb, offs, wfrag, out);
}
